// WindowAttention_38671885533444
// MI455X (gfx1250) — matrix operand order repaired and checked by reading, not compiled, not run on hardware
//
#include <hip/hip_runtime.h>
#include <hip/hip_bf16.h>

// ---------------------------------------------------------------------------
// Fused window attention for MI455X (gfx1250, wave32, WMMA f16->f32).
//   B=32, H=64, W=64, C=256, HEADS=8, HEAD_DIM=32, INNER=256
// One 256-thread workgroup per (batch, h-row); attention runs along W=64.
// Weights are pre-packed (fp32 -> f16, WMMA B-operand lane layout) into d_ws.
// ---------------------------------------------------------------------------

typedef __attribute__((ext_vector_type(16))) _Float16 v16h;
typedef __attribute__((ext_vector_type(8)))  _Float16 v8h;
typedef __attribute__((ext_vector_type(4)))  _Float16 v4h;
typedef __attribute__((ext_vector_type(8)))  float    v8f;

#define WMMA_F16(a, b, c) \
  __builtin_amdgcn_wmma_f32_16x16x32_f16(false, (a), false, (b), (short)0, (c), false, false)

// ---- per-lane operand gathers (contiguous k order, wave32) -----------------

// A operand (16x32 f16): lane<16 -> row M=lane, halves = K k0+0..15
// contiguous; lane>=16 -> row M=lane-16, halves = K k0+16..31.
__device__ __forceinline__ v16h load_A16(const _Float16* base, int ld, int row0,
                                         int k0, int lane) {
  const _Float16* p = base + (row0 + (lane & 15)) * ld + k0 + ((lane & 16) ? 16 : 0);
  union { v16h v; v8h h[2]; } u;
  u.h[0] = *(const v8h*)(p);
  u.h[1] = *(const v8h*)(p + 8);
  return u.v;
}

// B operand (32x16 f16) gathered from an LDS matrix stored as mat[col][k]
// with leading dim ld: lane<16 -> col N=lane, halves = K 0..15 contiguous;
// lane>=16 -> col N=lane-16, halves = K 16..31.
__device__ __forceinline__ v16h load_B16(const _Float16* base, int ld, int col0,
                                         int k0, int lane) {
  const _Float16* p = base + (col0 + (lane & 15)) * ld + k0 + ((lane & 16) ? 16 : 0);
  union { v16h v; v8h h[2]; } u;
  u.h[0] = *(const v8h*)(p);
  u.h[1] = *(const v8h*)(p + 8);
  return u.v;
}

// B operand from a pre-packed global tile: 32 lanes x 16 halves, contiguous.
__device__ __forceinline__ v16h load_Bpk(const _Float16* tile, int lane) {
  const _Float16* p = tile + lane * 16;
  union { v16h v; v8h h[2]; } u;
  u.h[0] = *(const v8h*)(p);
  u.h[1] = *(const v8h*)(p + 8);
  return u.v;
}

// D (16x16 f32 accum) -> LDS f16, row-major dst[m*ld + n].
// VGPR r: lane<16 -> (M=r, N=lane); lane>=16 -> (M=r+8, N=lane-16).
__device__ __forceinline__ void store_D16(v8f acc, _Float16* base, int ld,
                                          int row0, int col0, int lane) {
  int m0 = row0 + ((lane & 16) ? 8 : 0);
  int n  = col0 + (lane & 15);
#pragma unroll
  for (int r = 0; r < 8; ++r) base[(m0 + r) * ld + n] = (_Float16)acc[r];
}

// D -> LDS f16, transposed dst[n*ldT + m] (used to store v as v^T).
__device__ __forceinline__ void store_D16T(v8f acc, _Float16* base, int ldT,
                                           int row0, int col0, int lane) {
  int m0 = row0 + ((lane & 16) ? 8 : 0);
  int n  = col0 + (lane & 15);
#pragma unroll
  for (int r = 0; r < 8; ++r) base[n * ldT + (m0 + r)] = (_Float16)acc[r];
}

// D -> LDS f32 scores Sb[m*64 + n].
__device__ __forceinline__ void store_Df32(v8f acc, float* base, int row0,
                                           int col0, int lane) {
  int m0 = row0 + ((lane & 16) ? 8 : 0);
  int n  = col0 + (lane & 15);
#pragma unroll
  for (int r = 0; r < 8; ++r) base[(m0 + r) * 64 + n] = acc[r];
}

// ---------------------------------------------------------------------------
// Pack wq/wk/wv/wo (each [256 K x 256 N] fp32, row-major) into f16 WMMA
// B-operand tiles: tile (kt, nt) covers K = kt*32..+31, N = nt*16..+15.
// Tile storage: 512 halves = lane*16 contiguous halves per lane.
// dst layout: weight w -> wpk + w*65536, tile -> + (kt*16 + nt)*512.
// Grid: 512 blocks (4 weights x 128 tiles) of 32 threads.
// ---------------------------------------------------------------------------
__global__ void wa_pack_weights(const float* __restrict__ wq,
                                const float* __restrict__ wk,
                                const float* __restrict__ wv,
                                const float* __restrict__ wo,
                                _Float16* __restrict__ wpk) {
  int bid = blockIdx.x;
  int w   = bid >> 7;       // 0..3
  int t   = bid & 127;      // tile: kt*16 + nt
  int kt  = t >> 4, nt = t & 15;
  const float* src = (w == 0) ? wq : (w == 1) ? wk : (w == 2) ? wv : wo;
  _Float16* dst = wpk + (size_t)w * 65536 + (size_t)t * 512;

  int lane  = threadIdx.x & 31;
  int n     = nt * 16 + (lane & 15);
  int kbase = kt * 32 + ((lane & 16) ? 16 : 0);

  _Float16 tmp[16];
#pragma unroll
  for (int j = 0; j < 16; ++j) tmp[j] = (_Float16)src[(size_t)(kbase + j) * 256 + n];
  v8h* o = (v8h*)(dst + lane * 16);
  o[0] = *(v8h*)&tmp[0];
  o[1] = *(v8h*)&tmp[8];
}

// ---------------------------------------------------------------------------
// Main fused kernel. grid = 2048 (= 32 batches * 64 h-rows), block = 256.
// ---------------------------------------------------------------------------
__launch_bounds__(256, 1)
__global__ void wa_attn_fused(const float* __restrict__ x,
                              const _Float16* __restrict__ wpk,
                              const float* __restrict__ bo,
                              float* __restrict__ out) {
  constexpr float kScale = 0.17677669529663689f;  // 1/sqrt(32)

  __shared__ alignas(16) _Float16 xs [64 * 256];  // x row, f16        32 KB
  __shared__ alignas(16) _Float16 qh [64 * 32];   // q_head [pos][d]    4 KB
  __shared__ alignas(16) _Float16 kh [64 * 32];   // k_head [pos][d]    4 KB
  __shared__ alignas(16) _Float16 vts[32 * 64];   // v_head^T [d][pos]  4 KB
  __shared__ alignas(16) float    Sb [64 * 64];   // scores f32        16 KB
  __shared__ alignas(16) _Float16 Oh [64 * 32];   // head out [pos][d]  4 KB

  const int tid  = threadIdx.x;
  const int lane = tid & 31;
  const int wv   = tid >> 5;          // wave id 0..7
  const size_t row_base = (size_t)blockIdx.x * (64 * 256);

  // ---- stage 0: x[b, hrow, :, :] (64x256 f32) -> LDS f16, coalesced -------
  {
    const float4* xv = (const float4*)(x + row_base);
#pragma unroll
    for (int i = 0; i < 16; ++i) {
      int idx = i * 256 + tid;        // float4 index, 4096 total
      float4 f = xv[idx];
      v4h h4 = { (_Float16)f.x, (_Float16)f.y, (_Float16)f.z, (_Float16)f.w };
      *(v4h*)(xs + (size_t)idx * 4) = h4;
    }
  }
  __syncthreads();

  // persistent f32 accumulators: wave wv owns out tiles (mt = wv>>1,
  // nt = (wv&1)*8 + i) for i = 0..7 -> the full 64x256 block output.
  v8f cacc[8];
#pragma unroll
  for (int i = 0; i < 8; ++i) cacc[i] = (v8f){0.f, 0.f, 0.f, 0.f, 0.f, 0.f, 0.f, 0.f};

  _Float16* Pb = (_Float16*)Sb;            // f16 probs overlay (2-phase safe)
  const _Float16* wobase = wpk + (size_t)3 * 65536;

  for (int hd = 0; hd < 8; ++hd) {
    __syncthreads();  // protect qh/kh/vts/Oh from previous iteration readers

    // ---- QKV projection -------------------------------------------------
    // Wave wv owns the same 16x16 output tile (mt = wv>>1, nt = wv&1) for
    // q, k and v -> the 8 A-operands from xs are shared across all three.
    {
      const int mt = wv >> 1, nt = wv & 1;
      const int ntg = hd * 2 + nt;         // global N tile in [0,16)

      // prefetch next head's weight tiles toward the WGP (global_prefetch_b8)
      if (hd < 7) {
        __builtin_prefetch(wpk + (size_t)(wv * 16 + (hd + 1) * 2) * 512, 0, 1);
        __builtin_prefetch(wobase + (size_t)((hd + 1) * 16 + wv * 2) * 512, 0, 1);
      }

      v16h aop[8];
#pragma unroll
      for (int kt = 0; kt < 8; ++kt) aop[kt] = load_A16(xs, 256, mt * 16, kt * 32, lane);

#pragma unroll
      for (int mat = 0; mat < 3; ++mat) {
        const _Float16* wbase = wpk + (size_t)mat * 65536;
        v8f acc = {0.f, 0.f, 0.f, 0.f, 0.f, 0.f, 0.f, 0.f};
#pragma unroll
        for (int kt = 0; kt < 8; ++kt) {
          v16h b = load_Bpk(wbase + (size_t)(kt * 16 + ntg) * 512, lane);
          acc = WMMA_F16(aop[kt], b, acc);
        }
        if      (mat == 0) store_D16 (acc, qh, 32, mt * 16, nt * 16, lane);
        else if (mat == 1) store_D16 (acc, kh, 32, mt * 16, nt * 16, lane);
        else               store_D16T(acc, vts, 64, mt * 16, nt * 16, lane);
      }
    }
    __syncthreads();

    // ---- S = q . k^T : wave wv owns tiles (mt, nt) and (mt+2, nt) with
    // mt = wv>>2, nt = wv&3 -> shared B operand, two WMMAs. -----------------
    {
      const int mt = wv >> 2, nt = wv & 3;
      v16h b = load_B16(kh, 32, nt * 16, 0, lane);
#pragma unroll
      for (int i = 0; i < 2; ++i) {
        int m = (mt + 2 * i) * 16;
        v16h a = load_A16(qh, 32, m, 0, lane);
        v8f acc = {0.f, 0.f, 0.f, 0.f, 0.f, 0.f, 0.f, 0.f};
        acc = WMMA_F16(a, b, acc);
        store_Df32(acc, Sb, m, nt * 16, lane);
      }
    }
    __syncthreads();

    // ---- softmax: 4 lanes per row, 16 cols each; wave-local shuffles -----
    // (scale folded into the read; 2-phase so the f16 overlay is race-free)
    {
      const int r  = tid >> 2;             // row 0..63
      const int cq = (tid & 3) * 16;       // column quarter
      float e[16];
      float mx = -3.0e38f;
#pragma unroll
      for (int j = 0; j < 16; ++j) {
        e[j] = Sb[r * 64 + cq + j] * kScale;
        mx = fmaxf(mx, e[j]);
      }
      mx = fmaxf(mx, __shfl_xor(mx, 1));
      mx = fmaxf(mx, __shfl_xor(mx, 2));
      float sum = 0.f;
#pragma unroll
      for (int j = 0; j < 16; ++j) { e[j] = __expf(e[j] - mx); sum += e[j]; }
      sum += __shfl_xor(sum, 1);
      sum += __shfl_xor(sum, 2);
      float inv = 1.f / sum;
      __syncthreads();                     // all f32 reads of Sb done
#pragma unroll
      for (int j = 0; j < 16; ++j) Pb[r * 64 + cq + j] = (_Float16)(e[j] * inv);
    }
    __syncthreads();

    // ---- O = P . v : 1 tile per wave, K=64 (2 WMMA steps) ----------------
    {
      const int mt = wv >> 1, nt = wv & 1;
      v8f acc = {0.f, 0.f, 0.f, 0.f, 0.f, 0.f, 0.f, 0.f};
#pragma unroll
      for (int kt = 0; kt < 2; ++kt) {
        v16h a = load_A16(Pb, 64, mt * 16, kt * 32, lane);
        v16h b = load_B16(vts, 64, nt * 16, kt * 32, lane);
        acc = WMMA_F16(a, b, acc);
      }
      store_D16(acc, Oh, 32, mt * 16, nt * 16, lane);
    }
    __syncthreads();

    // ---- out += O_head . wo[head rows] : 8 tiles per wave, shared A ------
    {
      const int mt = wv >> 1;              // constant per wave
      v16h a = load_A16(Oh, 32, mt * 16, 0, lane);
#pragma unroll
      for (int i = 0; i < 8; ++i) {
        int nt = (wv & 1) * 8 + i;
        v16h b = load_Bpk(wobase + (size_t)(hd * 16 + nt) * 512, lane);
        cacc[i] = WMMA_F16(a, b, cacc[i]);
      }
    }
    // loop-top __syncthreads protects Oh/qh/kh/vts reuse
  }

  // ---- epilogue: add bias, store 64x256 f32 row ---------------------------
  float* orow = out + row_base;
  const int mt = wv >> 1;
  const int m0 = mt * 16 + ((lane & 16) ? 8 : 0);
#pragma unroll
  for (int i = 0; i < 8; ++i) {
    int nt = (wv & 1) * 8 + i;
    int n  = nt * 16 + (lane & 15);
    float bias = bo[n];
#pragma unroll
    for (int r = 0; r < 8; ++r) orow[(size_t)(m0 + r) * 256 + n] = cacc[i][r] + bias;
  }
}

// ---------------------------------------------------------------------------
extern "C" void kernel_launch(void* const* d_in, const int* in_sizes, int n_in,
                              void* d_out, int out_size, void* d_ws, size_t ws_size,
                              hipStream_t stream) {
  const float* x  = (const float*)d_in[0];
  const float* wq = (const float*)d_in[1];
  const float* wk = (const float*)d_in[2];
  const float* wv = (const float*)d_in[3];
  const float* wo = (const float*)d_in[4];
  const float* bo = (const float*)d_in[5];
  float* out = (float*)d_out;
  _Float16* wpk = (_Float16*)d_ws;    // 4 * 256*256 f16 = 512 KB

  // 1) pack fp32 weights into WMMA B-operand f16 tiles (L2-resident)
  wa_pack_weights<<<dim3(512), dim3(32), 0, stream>>>(wq, wk, wv, wo, wpk);
  // 2) fused QKV + attention + output projection, one block per (b, h-row)
  wa_attn_fused<<<dim3(32 * 64), dim3(256), 0, stream>>>(x, wpk, bo, out);
}
